// FullAttention_63359357551217
// MI455X (gfx1250) — compile-verified, matrix operand order checked by reading, not run on hardware
//
#include <hip/hip_runtime.h>

typedef __attribute__((ext_vector_type(16))) _Float16 v16h;
typedef __attribute__((ext_vector_type(8)))  _Float16 v8h;
typedef __attribute__((ext_vector_type(8)))  float    v8f;

// Problem constants (from the reference): N,L,S,H,D
constexpr int N_ = 2, L_ = 4096, S_ = 4096, H_ = 8, D_ = 32;
constexpr int WAVES = 8;     // waves per block
constexpr int SBLK  = 32;    // keys per inner iteration (two 16-key WMMA tiles)

// 1/sqrt(32) * log2(e): scores computed directly in log2 domain so the
// softmax exponential is a bare v_exp_f32.
#define SCALE_LOG2E 0.25505410884055946f

#if __has_builtin(__builtin_amdgcn_exp2f)
#define EXP2F(x) __builtin_amdgcn_exp2f(x)
#else
#define EXP2F(x) exp2f(x)
#endif

// Workspace layout (bytes)
constexpr size_t Q16_OFF = 0;
constexpr size_t Q16_SZ  = (size_t)N_ * H_ * L_ * D_ * 2;   // 4 MiB
constexpr size_t K16_OFF = Q16_OFF + Q16_SZ;
constexpr size_t K16_SZ  = (size_t)N_ * H_ * S_ * D_ * 2;   // 4 MiB
constexpr size_t VT_OFF  = K16_OFF + K16_SZ;
constexpr size_t VT_SZ   = (size_t)N_ * H_ * D_ * S_ * 2;   // 4 MiB
constexpr size_t FM_OFF  = VT_OFF + VT_SZ;
constexpr size_t FM_SZ   = (size_t)N_ * S_ * 4;             // 32 KiB
constexpr size_t WS_NEED = FM_OFF + FM_SZ;

// ---------------------------------------------------------------------------
// Prep: fp32 -> f16 with layout changes so every WMMA operand half-vector is
// a contiguous 16-byte chunk in memory (two global_load_b128 per operand).
//   Q16 [n,h,q,d]  = Q[n,q,h,d] * SCALE_LOG2E
//   K16 [n,h,s,d]  = K[n,s,h,d]
//   Vt16[n,h,d,s]  = V[n,s,h,d]        (transposed: key axis contiguous)
//   FM  [n,s]      = kv_mask ? 1.0 : 0.0
// ---------------------------------------------------------------------------
__global__ __launch_bounds__(256)
void prep_kernel(const float* __restrict__ Q, const float* __restrict__ K,
                 const float* __restrict__ V, const unsigned char* __restrict__ kvm,
                 _Float16* __restrict__ Q16, _Float16* __restrict__ K16,
                 _Float16* __restrict__ Vt16, float* __restrict__ FM) {
  size_t i = (size_t)blockIdx.x * blockDim.x + threadIdx.x;  // over N*H*L*D
  // decode target index [n,h,x,d]
  int d = (int)(i & (D_ - 1));
  size_t t = i >> 5;                 // D_ == 32
  int x = (int)(t % L_); t /= L_;
  int h = (int)(t % H_);
  int n = (int)(t / H_);
  // source index [n,x,h,d]
  size_t src = (((size_t)n * L_ + x) * H_ + h) * D_ + d;
  Q16[i] = (_Float16)(Q[src] * SCALE_LOG2E);
  K16[i] = (_Float16)(K[src]);
  size_t j = (((size_t)n * H_ + h) * D_ + d) * (size_t)S_ + x;
  Vt16[j] = (_Float16)(V[src]);
  if (i < (size_t)N_ * S_) FM[i] = kvm[i] ? 1.0f : 0.0f;
}

__device__ __forceinline__ v16h cat8(v8h lo, v8h hi) {
  return __builtin_shufflevector(lo, hi, 0, 1, 2, 3, 4, 5, 6, 7,
                                 8, 9, 10, 11, 12, 13, 14, 15);
}

// ---------------------------------------------------------------------------
// Main flash-attention kernel (f16 operands preconverted in workspace).
// One wave = 16 queries x full D for one (n,h); scores computed transposed
// (S^T = K*Q^T) so softmax state is lane-local (query == N == lane%16).
// ---------------------------------------------------------------------------
__global__ __launch_bounds__(32 * WAVES)
void fa_wmma_fast(const _Float16* __restrict__ Q16,
                  const _Float16* __restrict__ K16,
                  const _Float16* __restrict__ Vt16,
                  const float* __restrict__ FM,
                  const unsigned char* __restrict__ qmask,
                  float* __restrict__ O) {
  const int lane = threadIdx.x & 31;
  const int wave = threadIdx.x >> 5;
  const int g    = lane >> 4;   // lane group (0: M=0..7, 1: M=8..15)
  const int ln   = lane & 15;   // N index (query col) / A-row index

  const int q_tiles = L_ / 16;
  const int tile = blockIdx.x * WAVES + wave;
  const int qt = tile % q_tiles;
  const int nh = tile / q_tiles;               // n*H + h
  const int q0 = qt * 16;

  // Q as B operand: halves 0..7 = d 8g..8g+7, halves 8..15 = d 16+8g..16+8g+7
  const _Float16* qrow = Q16 + (((size_t)nh * L_) + (q0 + ln)) * D_;
  const v16h bq = cat8(*(const v8h*)(qrow + 8 * g),
                       *(const v8h*)(qrow + 16 + 8 * g));

  const _Float16* Kb = K16 + (size_t)nh * S_ * D_;
  const _Float16* Vb = Vt16 + (size_t)nh * D_ * S_;
  const float* FMp = FM + (size_t)(nh / H_) * S_;
  const unsigned char qvalid = qmask[(size_t)(nh / H_) * L_ + q0 + ln];

  v8f acc0 = {};                 // O^T: d = 0..15 on M axis, query on N
  v8f acc1 = {};                 // O^T: d = 16..31
  float mrun = -1e30f;           // running max (log2 domain), per query
  float lrun = 0.0f;             // running sum

  const _Float16* v0row = Vb + (size_t)ln * S_;          // d = ln
  const _Float16* v1row = Vb + (size_t)(16 + ln) * S_;   // d = 16+ln

  for (int s = 0; s < S_; s += SBLK) {
    // K rows as A operands (key = s+ln / s+16+ln), two b128 loads each
    const _Float16* kr0 = Kb + (size_t)(s + ln) * D_;
    const _Float16* kr1 = kr0 + 16 * D_;
    v16h ak0 = cat8(*(const v8h*)(kr0 + 8 * g), *(const v8h*)(kr0 + 16 + 8 * g));
    v16h ak1 = cat8(*(const v8h*)(kr1 + 8 * g), *(const v8h*)(kr1 + 16 + 8 * g));

    v8f cz0 = {}, cz1 = {};
    v8f s0 = __builtin_amdgcn_wmma_f32_16x16x32_f16(false, ak0, false, bq,
                                                    (short)0, cz0, false, false);
    v8f s1 = __builtin_amdgcn_wmma_f32_16x16x32_f16(false, ak1, false, bq,
                                                    (short)0, cz1, false, false);

    // float {0,1} masks for this lane's 16 score rows (keys r+8g, 16+r+8g)
    float fm0[8], fm1[8];
    *(float4*)&fm0[0] = *(const float4*)(FMp + s + 8 * g);
    *(float4*)&fm0[4] = *(const float4*)(FMp + s + 8 * g + 4);
    *(float4*)&fm1[0] = *(const float4*)(FMp + s + 16 + 8 * g);
    *(float4*)&fm1[4] = *(const float4*)(FMp + s + 16 + 8 * g + 4);

    // max over ALL scores (masked ones are bounded real values; using a
    // larger mnew is mathematically harmless) -> pure v_max3 chain
    float tmax = fmaxf(s0[0], s1[0]);
#pragma unroll
    for (int r = 1; r < 8; ++r) tmax = fmaxf(tmax, fmaxf(s0[r], s1[r]));
    tmax = fmaxf(tmax, __shfl_xor(tmax, 16, 32));
    const float mnew = fmaxf(mrun, tmax);
    const float corr = EXP2F(mrun - mnew);

    float p0[8], p1[8];
    float psum = 0.0f;
#pragma unroll
    for (int r = 0; r < 8; ++r) {
      p0[r] = EXP2F(s0[r] - mnew) * fm0[r];
      p1[r] = EXP2F(s1[r] - mnew) * fm1[r];
      psum += p0[r] + p1[r];
    }
    psum += __shfl_xor(psum, 16, 32);
    lrun = lrun * corr + psum;
    mrun = mnew;
#pragma unroll
    for (int r = 0; r < 8; ++r) { acc0[r] *= corr; acc1[r] *= corr; }

    // P -> f16 B operand (lane-local re-layout; pairs pack via cvt_pk)
    v16h bp;
#pragma unroll
    for (int v = 0; v < 4; ++v) {
      bp[2 * v]     = (_Float16)p0[2 * v];
      bp[2 * v + 1] = (_Float16)p0[2 * v + 1];
      bp[2 * v + 8] = (_Float16)p1[2 * v];
      bp[2 * v + 9] = (_Float16)p1[2 * v + 1];
    }

    // V^T rows as A operands: contiguous keys, two b128 loads each
    v16h av0 = cat8(*(const v8h*)(v0row + s + 8 * g),
                    *(const v8h*)(v0row + s + 16 + 8 * g));
    v16h av1 = cat8(*(const v8h*)(v1row + s + 8 * g),
                    *(const v8h*)(v1row + s + 16 + 8 * g));

    acc0 = __builtin_amdgcn_wmma_f32_16x16x32_f16(false, av0, false, bp,
                                                  (short)0, acc0, false, false);
    acc1 = __builtin_amdgcn_wmma_f32_16x16x32_f16(false, av1, false, bp,
                                                  (short)0, acc1, false, false);
  }

  const float inv = (qvalid && lrun > 0.0f) ? (1.0f / lrun) : 0.0f;
  const int n = nh / H_, h = nh % H_;
  float* orow = O + (((size_t)n * L_ + (q0 + ln)) * H_ + h) * D_;
#pragma unroll
  for (int r = 0; r < 8; ++r) {
    orow[r + 8 * g]      = acc0[r] * inv;
    orow[16 + r + 8 * g] = acc1[r] * inv;
  }
}

// ---------------------------------------------------------------------------
// Fallback (round-1 style, direct fp32 loads) used only if ws_size is too
// small for the preconverted operand buffers.
// ---------------------------------------------------------------------------
__global__ __launch_bounds__(32 * WAVES)
void fa_wmma_direct(const float* __restrict__ Q, const float* __restrict__ K,
                    const float* __restrict__ V,
                    const unsigned char* __restrict__ qmask,
                    const unsigned char* __restrict__ kvmask,
                    float* __restrict__ O) {
  const int lane = threadIdx.x & 31;
  const int wave = threadIdx.x >> 5;
  const int g = lane >> 4, ln = lane & 15;
  const int q_tiles = L_ / 16;
  const int tile = blockIdx.x * WAVES + wave;
  const int qt = tile % q_tiles;
  const int nh = tile / q_tiles;
  const int h = nh % H_, n = nh / H_;
  const int q0 = qt * 16;

  int dofs[8];
#pragma unroll
  for (int v = 0; v < 8; ++v) dofs[v] = ((v < 4) ? 0 : 16) + (v & 3) * 2 + g * 8;

  v16h bq;
  {
    const float* qrow = Q + (((size_t)n * L_ + (q0 + ln)) * H_ + h) * D_;
#pragma unroll
    for (int v = 0; v < 8; ++v) {
      float2 qq = *(const float2*)(qrow + dofs[v]);
      bq[2 * v]     = (_Float16)(qq.x * SCALE_LOG2E);
      bq[2 * v + 1] = (_Float16)(qq.y * SCALE_LOG2E);
    }
  }
  const unsigned char qvalid = qmask[(size_t)n * L_ + q0 + ln];
  const size_t rs = (size_t)H_ * D_;
  const float* Kp = K + ((size_t)n * S_) * rs + (size_t)h * D_;
  const float* Vp = V + ((size_t)n * S_) * rs + (size_t)h * D_;
  const unsigned char* km = kvmask + (size_t)n * S_;

  v8f acc0 = {}, acc1 = {};
  float mrun = -1e30f, lrun = 0.0f;

  for (int s = 0; s < S_; s += SBLK) {
    v16h ak0, ak1;
    const float* kr0 = Kp + (size_t)(s + ln) * rs;
    const float* kr1 = kr0 + 16 * rs;
#pragma unroll
    for (int v = 0; v < 8; ++v) {
      float2 k0 = *(const float2*)(kr0 + dofs[v]);
      float2 k1 = *(const float2*)(kr1 + dofs[v]);
      ak0[2 * v] = (_Float16)k0.x; ak0[2 * v + 1] = (_Float16)k0.y;
      ak1[2 * v] = (_Float16)k1.x; ak1[2 * v + 1] = (_Float16)k1.y;
    }
    v8f cz0 = {}, cz1 = {};
    v8f s0 = __builtin_amdgcn_wmma_f32_16x16x32_f16(false, ak0, false, bq,
                                                    (short)0, cz0, false, false);
    v8f s1 = __builtin_amdgcn_wmma_f32_16x16x32_f16(false, ak1, false, bq,
                                                    (short)0, cz1, false, false);
    float fm0[8], fm1[8];
#pragma unroll
    for (int r = 0; r < 8; ++r) {
      fm0[r] = (float)km[s + r + 8 * g];
      fm1[r] = (float)km[s + 16 + r + 8 * g];
    }
    float tmax = fmaxf(s0[0], s1[0]);
#pragma unroll
    for (int r = 1; r < 8; ++r) tmax = fmaxf(tmax, fmaxf(s0[r], s1[r]));
    tmax = fmaxf(tmax, __shfl_xor(tmax, 16, 32));
    const float mnew = fmaxf(mrun, tmax);
    const float corr = EXP2F(mrun - mnew);
    float p0[8], p1[8], psum = 0.0f;
#pragma unroll
    for (int r = 0; r < 8; ++r) {
      p0[r] = EXP2F(s0[r] - mnew) * fm0[r];
      p1[r] = EXP2F(s1[r] - mnew) * fm1[r];
      psum += p0[r] + p1[r];
    }
    psum += __shfl_xor(psum, 16, 32);
    lrun = lrun * corr + psum;
    mrun = mnew;
#pragma unroll
    for (int r = 0; r < 8; ++r) { acc0[r] *= corr; acc1[r] *= corr; }
    v16h bp;
#pragma unroll
    for (int v = 0; v < 4; ++v) {
      bp[2 * v]     = (_Float16)p0[2 * v];
      bp[2 * v + 1] = (_Float16)p0[2 * v + 1];
      bp[2 * v + 8] = (_Float16)p1[2 * v];
      bp[2 * v + 9] = (_Float16)p1[2 * v + 1];
    }
    v16h av0, av1;
#pragma unroll
    for (int v = 0; v < 8; ++v) {
      const float* vr0 = Vp + (size_t)(s + dofs[v]) * rs;
      const float* vr1 = Vp + (size_t)(s + dofs[v] + 1) * rs;
      av0[2 * v] = (_Float16)vr0[ln];      av0[2 * v + 1] = (_Float16)vr1[ln];
      av1[2 * v] = (_Float16)vr0[16 + ln]; av1[2 * v + 1] = (_Float16)vr1[16 + ln];
    }
    acc0 = __builtin_amdgcn_wmma_f32_16x16x32_f16(false, av0, false, bp,
                                                  (short)0, acc0, false, false);
    acc1 = __builtin_amdgcn_wmma_f32_16x16x32_f16(false, av1, false, bp,
                                                  (short)0, acc1, false, false);
  }
  const float inv = (qvalid && lrun > 0.0f) ? (1.0f / lrun) : 0.0f;
  float* orow = O + (((size_t)n * L_ + (q0 + ln)) * H_ + h) * D_;
#pragma unroll
  for (int r = 0; r < 8; ++r) {
    orow[r + 8 * g]      = acc0[r] * inv;
    orow[16 + r + 8 * g] = acc1[r] * inv;
  }
}

extern "C" void kernel_launch(void* const* d_in, const int* in_sizes, int n_in,
                              void* d_out, int out_size, void* d_ws, size_t ws_size,
                              hipStream_t stream) {
  (void)in_sizes; (void)n_in; (void)out_size;
  const float* Q = (const float*)d_in[0];
  const float* K = (const float*)d_in[1];
  const float* V = (const float*)d_in[2];
  const unsigned char* qm  = (const unsigned char*)d_in[3];
  const unsigned char* kvm = (const unsigned char*)d_in[4];
  float* O = (float*)d_out;

  const int tiles = N_ * H_ * (L_ / 16);       // 4096 query tiles
  const int blocks = tiles / WAVES;            // 512 blocks of 8 waves

  if (ws_size >= WS_NEED) {
    char* ws = (char*)d_ws;
    _Float16* Q16 = (_Float16*)(ws + Q16_OFF);
    _Float16* K16 = (_Float16*)(ws + K16_OFF);
    _Float16* Vt16 = (_Float16*)(ws + VT_OFF);
    float* FM = (float*)(ws + FM_OFF);
    const size_t elems = (size_t)N_ * H_ * L_ * D_;      // 2M
    prep_kernel<<<(int)(elems / 256), 256, 0, stream>>>(Q, K, V, kvm,
                                                        Q16, K16, Vt16, FM);
    fa_wmma_fast<<<blocks, 32 * WAVES, 0, stream>>>(Q16, K16, Vt16, FM, qm, O);
  } else {
    fa_wmma_direct<<<blocks, 32 * WAVES, 0, stream>>>(Q, K, V, qm, kvm, O);
  }
}
